// PhaseAttention_80255758893321
// MI455X (gfx1250) — matrix operand order repaired and checked by reading, not compiled, not run on hardware
//
#include <hip/hip_runtime.h>
#include <stdint.h>

// ---------------------------------------------------------------------------
// PhaseAttention for MI455X (gfx1250, wave32, WMMA + async-to-LDS staging)
// All heavy math runs through v_wmma_f32_16x16x32_bf16 (fp32 accumulate).
// Pipeline: cast -> QKV gemms -> rownorm -> V transpose -> dual logits gemm
//           -> softmax -> P@V gemm -> output projection gemm.
// The GEMMs stage operand tiles into LDS with global_load_async_to_lds_b128
// (ASYNCcnt-tracked DMA), double-buffered so the copy of slab k+1 overlaps
// the WMMAs of slab k. Logits (67MB fp32) stay L2-resident (192MB L2).
// ---------------------------------------------------------------------------

typedef __attribute__((ext_vector_type(16))) __bf16 v16bf;
typedef __attribute__((ext_vector_type(8)))  float  v8f;

union FragCvt { uint4 q[2]; v16bf v; };

__device__ __forceinline__ unsigned short f2bf(float f) {
  union { float f; unsigned u; } x; x.f = f;
  unsigned u = x.u;
  return (unsigned short)((u + 0x7FFFu + ((u >> 16) & 1u)) >> 16); // RNE
}
__device__ __forceinline__ float bf2f(unsigned short h) {
  union { unsigned u; float f; } x; x.u = ((unsigned)h) << 16;
  return x.f;
}

__device__ __forceinline__ v8f wmma_bf16(v16bf a, v16bf b, v8f c) {
  // (neg_a, A, neg_b, B, c_mod, C, reuse_a, reuse_b)
  return __builtin_amdgcn_wmma_f32_16x16x32_bf16(false, a, false, b,
                                                 (short)0, c, false, false);
}

// Async DMA: one 16B chunk global -> LDS, tracked by ASYNCcnt.
// vdst VGPR = per-lane LDS byte address, vaddr = 64-bit global address.
__device__ __forceinline__ void async_copy_b128(void* lds_ptr,
                                                const void* gptr) {
  unsigned loff = (unsigned)(size_t)lds_ptr;  // LDS aperture: addr[31:0]
  asm volatile("global_load_async_to_lds_b128 %0, %1, off"
               :
               : "v"(loff), "v"(gptr)
               : "memory");
}
#define WAIT_ASYNC0() asm volatile("s_wait_asynccnt 0x0" ::: "memory")

// ---------------------------------------------------------------------------
// Direct-from-global fragments (used by the logits kernel).
// A fragment (16x32 bf16, row-major source). ISA 7.12.2 layout:
// lane L: M = row0 + (L&15); VGPR0-3 hold K = k0 + (L>>4)*8 + 0..7,
// VGPR4-7 hold K = k0 + 16 + (L>>4)*8 + 0..7.
// ---------------------------------------------------------------------------
__device__ __forceinline__ v16bf load_a_frag(const unsigned short* A, int lda,
                                             int row0, int k0, int lane) {
  int half = lane >> 4;
  const unsigned short* p =
      A + (size_t)(row0 + (lane & 15)) * lda + k0 + half * 8;
  FragCvt t;
  t.q[0] = *(const uint4*)p;        // K halves 0..7 / 8..15
  t.q[1] = *(const uint4*)(p + 16); // K halves 16..23 / 24..31
  return t.v;
}

// B fragment: lane = N column (L&15), same K layout as A: K = k0 + (L>>4)*8
// + 0..7 and k0 + 16 + (L>>4)*8 + 0..7. B operand is stored N x K row-major
// (rows are columns of the logical B), so this is two 16-byte loads.
__device__ __forceinline__ v16bf load_b_frag(const unsigned short* Bm, int ldb,
                                             int col0, int k0, int lane) {
  const unsigned short* p =
      Bm + (size_t)(col0 + (lane & 15)) * ldb + k0 + (lane >> 4) * 8;
  FragCvt t;
  t.q[0] = *(const uint4*)p;
  t.q[1] = *(const uint4*)(p + 16);
  return t.v;
}

// ---------------------------------------------------------------------------
// LDS-staged fragments. LDS slabs are [rows][RS] bf16 with RS = 40 halfs
// (80B row pitch): bank stride 20 -> 16 consecutive rows touch 16 distinct
// banks (conflict-free fragment reads) and every 16B chunk stays 16B-aligned.
// ---------------------------------------------------------------------------
#define RS 40

__device__ __forceinline__ v16bf lds_a_frag(const unsigned short* sh, int row0,
                                            int lane) {
  const unsigned short* p = sh + (row0 + (lane & 15)) * RS + (lane >> 4) * 8;
  FragCvt t;
  t.q[0] = *(const uint4*)p;        // ds_load_b128
  t.q[1] = *(const uint4*)(p + 16);
  return t.v;
}
__device__ __forceinline__ v16bf lds_b_frag(const unsigned short* sh, int col0,
                                            int lane) {
  const unsigned short* p = sh + (col0 + (lane & 15)) * RS + (lane >> 4) * 8;
  FragCvt t;
  t.q[0] = *(const uint4*)p;
  t.q[1] = *(const uint4*)(p + 16);
  return t.v;
}

// ---------------------------------------------------------------------------
// fp32 -> bf16 cast
// ---------------------------------------------------------------------------
__global__ void cast_bf16_kernel(const float* __restrict__ src,
                                 unsigned short* __restrict__ dst, int n) {
  int i = blockIdx.x * blockDim.x + threadIdx.x;
  if (i < n) dst[i] = f2bf(src[i]);
}

// ---------------------------------------------------------------------------
// General GEMM: C = A(bf16, MxK row-major) * Bm(bf16, NxK row-major)^T
// optional fp32 bias per column; output fp32 or bf16. Batched via blockIdx.z.
// Block: 256 threads (8 waves). Wave tile 64x64 (4x4 of 16x16 WMMA tiles).
// WG tile: 128(M) x 256(N). Operands double-buffer-staged in LDS via
// global_load_async_to_lds_b128; copy of slab k+1 overlaps WMMA of slab k.
// ---------------------------------------------------------------------------
__global__ void __launch_bounds__(256)
gemm_bt_kernel(const unsigned short* __restrict__ A,
               const unsigned short* __restrict__ Bm,
               const float* __restrict__ bias, void* __restrict__ Cv,
               int lda, int ldb, int ldc, int K,
               long long strideA, long long strideB, long long strideC,
               int out_bf16) {
  __shared__ __attribute__((aligned(16))) unsigned short shA[2][128 * RS];
  __shared__ __attribute__((aligned(16))) unsigned short shB[2][256 * RS];

  const int lane = threadIdx.x & 31;
  const int wave = threadIdx.x >> 5;
  const int waveM = wave & 1;   // 2 waves along M
  const int waveN = wave >> 1;  // 4 waves along N

  A  += (size_t)blockIdx.z * strideA;
  Bm += (size_t)blockIdx.z * strideB;

  const int rowW = blockIdx.y * 128;  // WG tile origin
  const int colW = blockIdx.x * 256;

  // Stage one 32-k slab: A 128x32 (512 x 16B chunks), B 256x32 (1024 chunks).
  // 256 threads -> 2 A-chunks + 4 B-chunks per thread, all async DMA.
  auto issue_stage = [&](int par, int k0) {
#pragma unroll
    for (int i = 0; i < 2; ++i) {
      int c = threadIdx.x + i * 256;               // 0..511
      int r = c >> 2, cw = c & 3;
      const unsigned short* g =
          A + (size_t)(rowW + r) * lda + k0 + cw * 8;
      async_copy_b128(&shA[par][r * RS + cw * 8], g);
    }
#pragma unroll
    for (int i = 0; i < 4; ++i) {
      int c = threadIdx.x + i * 256;               // 0..1023
      int r = c >> 2, cw = c & 3;
      const unsigned short* g =
          Bm + (size_t)(colW + r) * ldb + k0 + cw * 8;
      async_copy_b128(&shB[par][r * RS + cw * 8], g);
    }
  };

  v8f zero = {0.f, 0.f, 0.f, 0.f, 0.f, 0.f, 0.f, 0.f};
  v8f acc[4][4];
#pragma unroll
  for (int i = 0; i < 4; ++i)
#pragma unroll
    for (int j = 0; j < 4; ++j) acc[i][j] = zero;

  const int row0 = waveM * 64;  // wave tile origin inside LDS slab
  const int col0 = waveN * 64;

  issue_stage(0, 0);
  WAIT_ASYNC0();
  __syncthreads();

  for (int k0 = 0; k0 < K; k0 += 32) {
    const int cur = (k0 >> 5) & 1;
    if (k0 + 32 < K) issue_stage(cur ^ 1, k0 + 32);  // overlap DMA w/ WMMA

    v16bf af[4], bfr[4];
#pragma unroll
    for (int i = 0; i < 4; ++i) af[i] = lds_a_frag(shA[cur], row0 + 16 * i, lane);
#pragma unroll
    for (int j = 0; j < 4; ++j) bfr[j] = lds_b_frag(shB[cur], col0 + 16 * j, lane);
#pragma unroll
    for (int i = 0; i < 4; ++i)
#pragma unroll
      for (int j = 0; j < 4; ++j) acc[i][j] = wmma_bf16(af[i], bfr[j], acc[i][j]);

    if (k0 + 32 < K) {
      WAIT_ASYNC0();     // my slab-(k+1) DMAs landed in LDS
      __syncthreads();   // everyone's landed; everyone done reading slab k
    }
  }

  // Epilogue. C/D layout: lane holds column n, rows m0 + r for VGPR r.
#pragma unroll
  for (int i = 0; i < 4; ++i) {
#pragma unroll
    for (int j = 0; j < 4; ++j) {
      const int n  = colW + col0 + 16 * j + (lane & 15);
      const int m0 = rowW + row0 + 16 * i + (lane >> 4) * 8;
      const float bv = bias ? bias[n] : 0.f;
      if (out_bf16) {
        unsigned short* C = (unsigned short*)Cv + (size_t)blockIdx.z * strideC;
#pragma unroll
        for (int r = 0; r < 8; ++r)
          C[(size_t)(m0 + r) * ldc + n] = f2bf(acc[i][j][r] + bv);
      } else {
        float* C = (float*)Cv + (size_t)blockIdx.z * strideC;
#pragma unroll
        for (int r = 0; r < 8; ++r)
          C[(size_t)(m0 + r) * ldc + n] = acc[i][j][r] + bv;
      }
    }
  }
}

// ---------------------------------------------------------------------------
// Dual logits GEMM: per batch computes S1 = Q*K^T and S2 = Qn*Kn^T with two
// accumulator sets in one k-sweep, epilogue L = cos(pi*S2) * (S1/sqrt(D)).
// Wave tile 32x32 (2x2 of 16x16, dual accum). WG tile 64(M) x 128(N).
// Direct global loads (4 streaming operands; clause-pipelined by compiler).
// ---------------------------------------------------------------------------
__global__ void __launch_bounds__(256)
logits_kernel(const unsigned short* __restrict__ Q,
              const unsigned short* __restrict__ Qn,
              const unsigned short* __restrict__ Kd,
              const unsigned short* __restrict__ Kn,
              float* __restrict__ L, int S, int D) {
  const int lane = threadIdx.x & 31;
  const int wave = threadIdx.x >> 5;
  const int waveM = wave & 1;
  const int waveN = wave >> 1;

  const size_t qoff = (size_t)blockIdx.z * S * D;
  const unsigned short* Qb  = Q  + qoff;
  const unsigned short* Qnb = Qn + qoff;
  const unsigned short* Kb  = Kd + qoff;
  const unsigned short* Knb = Kn + qoff;
  float* Lb = L + (size_t)blockIdx.z * S * S;

  const int row0 = blockIdx.y * 64  + waveM * 32;
  const int col0 = blockIdx.x * 128 + waveN * 32;

  v8f zero = {0.f, 0.f, 0.f, 0.f, 0.f, 0.f, 0.f, 0.f};
  v8f s1[2][2], s2[2][2];
#pragma unroll
  for (int i = 0; i < 2; ++i)
#pragma unroll
    for (int j = 0; j < 2; ++j) { s1[i][j] = zero; s2[i][j] = zero; }

  for (int k0 = 0; k0 < D; k0 += 32) {
    v16bf aq[2], an[2], bq[2], bn[2];
#pragma unroll
    for (int i = 0; i < 2; ++i) {
      aq[i] = load_a_frag(Qb,  D, row0 + 16 * i, k0, lane);
      an[i] = load_a_frag(Qnb, D, row0 + 16 * i, k0, lane);
    }
#pragma unroll
    for (int j = 0; j < 2; ++j) {
      bq[j] = load_b_frag(Kb,  D, col0 + 16 * j, k0, lane);
      bn[j] = load_b_frag(Knb, D, col0 + 16 * j, k0, lane);
    }
#pragma unroll
    for (int i = 0; i < 2; ++i)
#pragma unroll
      for (int j = 0; j < 2; ++j) {
        s1[i][j] = wmma_bf16(aq[i], bq[j], s1[i][j]);
        s2[i][j] = wmma_bf16(an[i], bn[j], s2[i][j]);
      }
  }

  const float scale = 0.03125f;           // 1/sqrt(1024)
  const float PI = 3.14159265358979f;
#pragma unroll
  for (int i = 0; i < 2; ++i)
#pragma unroll
    for (int j = 0; j < 2; ++j) {
      const int n  = col0 + 16 * j + (lane & 15);
      const int m0 = row0 + 16 * i + (lane >> 4) * 8;
#pragma unroll
      for (int r = 0; r < 8; ++r) {
        float amp = s1[i][j][r] * scale;
        float ph  = __cosf(PI * s2[i][j][r]);   // v_cos_f32
        Lb[(size_t)(m0 + r) * S + n] = ph * amp;
      }
    }
}

// ---------------------------------------------------------------------------
// Row L2-norm: Xn[row] = X[row] / (||X[row]|| + 1e-8), bf16 in/out.
// One 256-thread block per row of D=1024.
// ---------------------------------------------------------------------------
__global__ void __launch_bounds__(256)
rownorm_kernel(const unsigned short* __restrict__ X,
               unsigned short* __restrict__ Xn, int D) {
  const size_t base = (size_t)blockIdx.x * D;
  float v[4];
  float ss = 0.f;
#pragma unroll
  for (int i = 0; i < 4; ++i) {
    v[i] = bf2f(X[base + i * 256 + threadIdx.x]);
    ss += v[i] * v[i];
  }
#pragma unroll
  for (int off = 16; off > 0; off >>= 1) ss += __shfl_xor(ss, off, 32);
  __shared__ float sm[8];
  if ((threadIdx.x & 31) == 0) sm[threadIdx.x >> 5] = ss;
  __syncthreads();
  float tot = 0.f;
#pragma unroll
  for (int w = 0; w < 8; ++w) tot += sm[w];
  const float inv = 1.f / (sqrtf(tot) + 1e-8f);
#pragma unroll
  for (int i = 0; i < 4; ++i)
    Xn[base + i * 256 + threadIdx.x] = f2bf(v[i] * inv);
}

// ---------------------------------------------------------------------------
// bf16 matrix transpose via LDS tile (so the P@V gemm gets contiguous rows).
// src: [rows x cols] per batch -> dst: [cols x rows] per batch.
// ---------------------------------------------------------------------------
__global__ void __launch_bounds__(256)
transpose_kernel(const unsigned short* __restrict__ src,
                 unsigned short* __restrict__ dst, int rows, int cols) {
  __shared__ unsigned short tile[32][33];
  const size_t boff = (size_t)blockIdx.z * rows * cols;
  const int c0 = blockIdx.x * 32;
  const int r0 = blockIdx.y * 32;
  const int tx = threadIdx.x;   // 32
  const int ty = threadIdx.y;   // 8
#pragma unroll
  for (int i = 0; i < 4; ++i)
    tile[ty + 8 * i][tx] = src[boff + (size_t)(r0 + ty + 8 * i) * cols + c0 + tx];
  __syncthreads();
#pragma unroll
  for (int i = 0; i < 4; ++i)
    dst[boff + (size_t)(c0 + ty + 8 * i) * rows + r0 + tx] = tile[tx][ty + 8 * i];
}

// ---------------------------------------------------------------------------
// Row softmax over S=2048 fp32 logits -> bf16 probabilities.
// One 256-thread block per row; 8 elements per thread.
// ---------------------------------------------------------------------------
__global__ void __launch_bounds__(256)
softmax_kernel(const float* __restrict__ L, unsigned short* __restrict__ P,
               int S) {
  const size_t base = (size_t)blockIdx.x * S;
  float v[8];
  float m = -3.402823e38f;
#pragma unroll
  for (int i = 0; i < 8; ++i) {
    v[i] = L[base + i * 256 + threadIdx.x];
    m = fmaxf(m, v[i]);
  }
#pragma unroll
  for (int off = 16; off > 0; off >>= 1) m = fmaxf(m, __shfl_xor(m, off, 32));
  __shared__ float smx[8];
  __shared__ float sms[8];
  if ((threadIdx.x & 31) == 0) smx[threadIdx.x >> 5] = m;
  __syncthreads();
  float M = smx[0];
#pragma unroll
  for (int w = 1; w < 8; ++w) M = fmaxf(M, smx[w]);

  float s = 0.f;
#pragma unroll
  for (int i = 0; i < 8; ++i) {
    v[i] = __expf(v[i] - M);   // v_exp_f32
    s += v[i];
  }
#pragma unroll
  for (int off = 16; off > 0; off >>= 1) s += __shfl_xor(s, off, 32);
  if ((threadIdx.x & 31) == 0) sms[threadIdx.x >> 5] = s;
  __syncthreads();
  float tot = 0.f;
#pragma unroll
  for (int w = 0; w < 8; ++w) tot += sms[w];
  const float inv = 1.f / tot;
#pragma unroll
  for (int i = 0; i < 8; ++i)
    P[base + i * 256 + threadIdx.x] = f2bf(v[i] * inv);
}

// ---------------------------------------------------------------------------
// Host-side orchestration
// ---------------------------------------------------------------------------
extern "C" void kernel_launch(void* const* d_in, const int* in_sizes, int n_in,
                              void* d_out, int out_size, void* d_ws,
                              size_t ws_size, hipStream_t stream) {
  (void)in_sizes; (void)n_in; (void)out_size; (void)ws_size;

  const int Bn = 4, S = 2048, D = 1024;
  const int Mtot = Bn * S;                       // 8192
  const size_t XN = (size_t)Mtot * D;            // 8388608
  const size_t WN = (size_t)D * D;               // 1048576
  const size_t LN = (size_t)Bn * S * S;          // 16777216

  const float* x  = (const float*)d_in[0];
  const float* Wq = (const float*)d_in[1];
  const float* bq = (const float*)d_in[2];
  const float* Wk = (const float*)d_in[3];
  const float* bk = (const float*)d_in[4];
  const float* Wv = (const float*)d_in[5];
  const float* bv = (const float*)d_in[6];
  const float* Wo = (const float*)d_in[7];
  const float* bo = (const float*)d_in[8];

  // Workspace carve-up (256B aligned slabs).
  char* base = (char*)d_ws;
  size_t off = 0;
  auto alloc = [&](size_t bytes) {
    void* p = base + off;
    off += (bytes + 255) & ~(size_t)255;
    return p;
  };
  unsigned short* xb  = (unsigned short*)alloc(XN * 2);
  unsigned short* Wqb = (unsigned short*)alloc(WN * 2);
  unsigned short* Wkb = (unsigned short*)alloc(WN * 2);
  unsigned short* Wvb = (unsigned short*)alloc(WN * 2);
  unsigned short* Wob = (unsigned short*)alloc(WN * 2);
  unsigned short* Qb  = (unsigned short*)alloc(XN * 2);
  unsigned short* Kb  = (unsigned short*)alloc(XN * 2);
  unsigned short* Vb  = (unsigned short*)alloc(XN * 2);
  unsigned short* Qnb = (unsigned short*)alloc(XN * 2);
  unsigned short* Knb = (unsigned short*)alloc(XN * 2);
  unsigned short* Vt  = (unsigned short*)alloc(XN * 2);
  float*          Lg  = (float*)alloc(LN * 4);
  unsigned short* P   = (unsigned short*)alloc(LN * 2);
  unsigned short* AOb = (unsigned short*)alloc(XN * 2);

  // 1) casts to bf16
  cast_bf16_kernel<<<(XN + 255) / 256, 256, 0, stream>>>(x, xb, (int)XN);
  cast_bf16_kernel<<<(WN + 255) / 256, 256, 0, stream>>>(Wq, Wqb, (int)WN);
  cast_bf16_kernel<<<(WN + 255) / 256, 256, 0, stream>>>(Wk, Wkb, (int)WN);
  cast_bf16_kernel<<<(WN + 255) / 256, 256, 0, stream>>>(Wv, Wvb, (int)WN);
  cast_bf16_kernel<<<(WN + 255) / 256, 256, 0, stream>>>(Wo, Wob, (int)WN);

  // 2) QKV projections: Y = X * W^T + b   (out bf16)
  dim3 gQKV(D / 256, Mtot / 128, 1);
  gemm_bt_kernel<<<gQKV, 256, 0, stream>>>(xb, Wqb, bq, Qb, D, D, D, D, 0, 0, 0, 1);
  gemm_bt_kernel<<<gQKV, 256, 0, stream>>>(xb, Wkb, bk, Kb, D, D, D, D, 0, 0, 0, 1);
  gemm_bt_kernel<<<gQKV, 256, 0, stream>>>(xb, Wvb, bv, Vb, D, D, D, D, 0, 0, 0, 1);

  // 3) row norms -> Qn, Kn
  rownorm_kernel<<<Mtot, 256, 0, stream>>>(Qb, Qnb, D);
  rownorm_kernel<<<Mtot, 256, 0, stream>>>(Kb, Knb, D);

  // 4) V transpose per batch: Vt[b][d][k] = V[b][k][d]
  transpose_kernel<<<dim3(D / 32, S / 32, Bn), dim3(32, 8), 0, stream>>>(Vb, Vt, S, D);

  // 5) dual logits: L = cos(pi * Qn Kn^T) * (Q K^T / sqrt(D))
  logits_kernel<<<dim3(S / 128, S / 64, Bn), 256, 0, stream>>>(Qb, Qnb, Kb, Knb, Lg, S, D);

  // 6) softmax over keys -> P (bf16)
  softmax_kernel<<<Mtot, 256, 0, stream>>>(Lg, P, S);

  // 7) attention output: AO = P @ V   (B operand = rows of Vt)
  gemm_bt_kernel<<<dim3(D / 256, S / 128, Bn), 256, 0, stream>>>(
      P, Vt, nullptr, AOb, S, S, D, S,
      (long long)S * S, (long long)D * S, (long long)S * D, 1);

  // 8) output projection: out = AO * Wo^T + bo  (fp32)
  gemm_bt_kernel<<<gQKV, 256, 0, stream>>>(AOb, Wob, bo, d_out, D, D, D, D, 0, 0, 0, 0);
}
